// TransformerBlock_12799002542095
// MI455X (gfx1250) — matrix operand order repaired and checked by reading, not compiled, not run on hardware
//
#include <hip/hip_runtime.h>
#include <hip/hip_bf16.h>
#include <math.h>

// ---------------- problem constants ----------------
#define BB   2
#define CC   96
#define NHH  8
#define CHD  12          // CC / NHH
#define HHH  128
#define WWW  128
#define HW   (HHH*WWW)   // 16384
#define HIDQ 255         // HID
#define HID2 510         // 2*HID
#define HID4 1020        // 4*HID

// ---------------- WMMA types ----------------
typedef __attribute__((ext_vector_type(16))) short   v16s;
typedef __attribute__((ext_vector_type(16))) __bf16  v16bf;
typedef __attribute__((ext_vector_type(8)))  float   v8f;

__device__ __forceinline__ unsigned short f2bf(float f) {
  // round-to-nearest-even fp32 -> bf16
  unsigned u = __builtin_bit_cast(unsigned, f);
  u += 0x7fffu + ((u >> 16) & 1u);
  return (unsigned short)(u >> 16);
}

__device__ __forceinline__ unsigned pack2bf(float lo, float hi) {
  return (unsigned)f2bf(lo) | ((unsigned)f2bf(hi) << 16);
}

__device__ __forceinline__ v8f wmma_bf16(const v16s& as, const v16s& bs, v8f acc) {
  v16bf a = __builtin_bit_cast(v16bf, as);
  v16bf b = __builtin_bit_cast(v16bf, bs);
  return __builtin_amdgcn_wmma_f32_16x16x32_bf16(false, a, false, b, (short)0, acc,
                                                 false, false);
}

// ---------------- LayerNorm over channel axis (per pixel) ----------------
__global__ void ln_kernel(const float* __restrict__ x, const float* __restrict__ w,
                          const float* __restrict__ b, float* __restrict__ o) {
  int idx = blockIdx.x * blockDim.x + threadIdx.x;
  if (idx >= BB * HW) return;
  int bb = idx / HW, p = idx % HW;
  const float* xp = x + (size_t)bb * CC * HW + p;
  float s1 = 0.f, s2 = 0.f;
  for (int c = 0; c < CC; ++c) { float v = xp[(size_t)c * HW]; s1 += v; s2 += v * v; }
  float mu = s1 / CC;
  float var = s2 / CC - mu * mu;
  float r = rsqrtf(var + 1e-5f);
  float* op = o + (size_t)bb * CC * HW + p;
  for (int c = 0; c < CC; ++c)
    op[(size_t)c * HW] = (xp[(size_t)c * HW] - mu) * r * w[c] + b[c];
}

// ---------------- bf16 WMMA GEMM: C[M,N] = W[M,K]*X[K,N] (+Res), per-batch z ----
// 64x64 block tile, K-step 32. LDS tiles stored PRE-PERMUTED into WMMA lane
// fragment order so each lane reads its whole fragment as one 32B vector load.
//   A: lane = (m&15) + 16*((k>>3)&1) ; e = (k&7) + ((k&16)>>1)
//   B: lane = (n&15) + 16*((k>>3)&1) ; e = (k&7) + ((k&16)>>1)
// Even-k pairs land at even e in the same lane -> staged as packed b32 stores.
// All global loads are clamped+masked (branchless); epilogue guards are hoisted
// to block-uniform branches (no per-element exec manipulation).
#define GBM 64
#define GBN 64
#define GBK 32

__global__ __launch_bounds__(256)
void wmma_gemm_kernel(const float* __restrict__ Wt, const float* __restrict__ X,
                      const float* __restrict__ Res, float* __restrict__ Cout,
                      int M, int K, int N) {
  // [tile(4)][lane(32)][e(16)] ushort, 16B-aligned fragments
  __shared__ unsigned short lAp[4 * 32 * 16];
  __shared__ unsigned short lBp[4 * 32 * 16];
  int tid  = threadIdx.x;
  int lane = tid & 31, wave = tid >> 5;
  int wm   = wave >> 1, wn = wave & 1;     // 4 M-tiles x (2 waves x 2 sub) N-tiles
  int half = lane >> 4, lr = lane & 15;
  int m0 = blockIdx.y * GBM;
  int n0 = blockIdx.x * GBN;
  size_t zX = (size_t)blockIdx.z * K * N;
  size_t zC = (size_t)blockIdx.z * M * N;
  v8f acc0 = {0.f,0.f,0.f,0.f,0.f,0.f,0.f,0.f};
  v8f acc1 = {0.f,0.f,0.f,0.f,0.f,0.f,0.f,0.f};
  int nk = (K + GBK - 1) / GBK;
  for (int kt = 0; kt < nk; ++kt) {
    int k0 = kt * GBK;
    // ---- stage A: 64 rows x 16 k-pairs = 1024 items, 4 per thread ----
#pragma unroll
    for (int it = 0; it < 4; ++it) {
      int i  = tid + it * 256;             // i = r*16 + kp
      int r  = i >> 4;
      int k  = (i & 15) * 2;               // even k
      int m  = m0 + r;
      int mc = min(m, M - 1);
      int kg = k0 + k;
      int kc = min(kg, K - 2);             // K even for all call sites
      const float2 f = *(const float2*)(Wt + (size_t)mc * K + kc);
      float a0 = (m < M && kg     < K) ? f.x : 0.f;
      float a1 = (m < M && kg + 1 < K) ? f.y : 0.f;
      int ln = (r & 15) + 16 * ((k >> 3) & 1);
      int e  = (k & 7) + ((k & 16) >> 1);
      *(unsigned*)&lAp[(((r >> 4) * 32 + ln) << 4) + e] = pack2bf(a0, a1);
    }
    // ---- stage B: 64 cols x 16 k-pairs = 1024 items, 4 per thread ----
#pragma unroll
    for (int it = 0; it < 4; ++it) {
      int i  = tid + it * 256;             // i = kp*64 + n  (n fastest: coalesced)
      int n  = i & 63;
      int k  = (i >> 6) * 2;               // even k
      int kg = k0 + k;
      int kc0 = min(kg,     K - 1);
      int kc1 = min(kg + 1, K - 1);
      float b0 = X[zX + (size_t)kc0 * N + (n0 + n)];
      float b1 = X[zX + (size_t)kc1 * N + (n0 + n)];
      b0 = (kg     < K) ? b0 : 0.f;
      b1 = (kg + 1 < K) ? b1 : 0.f;
      int ln = (n & 15) + 16 * ((k >> 3) & 1);
      int e  = (k & 7) + ((k & 16) >> 1);
      *(unsigned*)&lBp[(((n >> 4) * 32 + ln) << 4) + e] = pack2bf(b0, b1);
    }
    __syncthreads();
    v16s a  = *(const v16s*)&lAp[((wm * 32 + lane) << 4)];
    v16s b0 = *(const v16s*)&lBp[(((wn * 2 + 0) * 32 + lane) << 4)];
    v16s b1 = *(const v16s*)&lBp[(((wn * 2 + 1) * 32 + lane) << 4)];
    acc0 = wmma_bf16(a, b0, acc0);
    acc1 = wmma_bf16(a, b1, acc1);
    __syncthreads();
  }
  // ---- epilogue: fused residual; all guards block-uniform ----
  int mbase = m0 + wm * 16 + (half ? 8 : 0);
  int nA = n0 + (wn * 2 + 0) * 16 + lr;
  int nB = n0 + (wn * 2 + 1) * 16 + lr;
  if (m0 + GBM <= M) {                     // interior: straight-line, coalesced
    if (Res) {
#pragma unroll
      for (int r = 0; r < 8; ++r) {
        size_t oa = zC + (size_t)(mbase + r) * N + nA;
        size_t ob = zC + (size_t)(mbase + r) * N + nB;
        Cout[oa] = acc0[r] + Res[oa];
        Cout[ob] = acc1[r] + Res[ob];
      }
    } else {
#pragma unroll
      for (int r = 0; r < 8; ++r) {
        Cout[zC + (size_t)(mbase + r) * N + nA] = acc0[r];
        Cout[zC + (size_t)(mbase + r) * N + nB] = acc1[r];
      }
    }
  } else {                                 // M-edge block: guarded
    if (Res) {
#pragma unroll
      for (int r = 0; r < 8; ++r) {
        int m = mbase + r;
        if (m < M) {
          size_t oa = zC + (size_t)m * N + nA;
          size_t ob = zC + (size_t)m * N + nB;
          Cout[oa] = acc0[r] + Res[oa];
          Cout[ob] = acc1[r] + Res[ob];
        }
      }
    } else {
#pragma unroll
      for (int r = 0; r < 8; ++r) {
        int m = mbase + r;
        if (m < M) {
          Cout[zC + (size_t)m * N + nA] = acc0[r];
          Cout[zC + (size_t)m * N + nB] = acc1[r];
        }
      }
    }
  }
}

// ---------------- generic grouped/depthwise 3x3, zero pad, optional relu --------
__global__ void dwconv_kernel(const float* __restrict__ in, const float* __restrict__ wgt,
                              float* __restrict__ out, int Cout, int mult, int relu) {
  size_t idx = (size_t)blockIdx.x * blockDim.x + threadIdx.x;
  size_t total = (size_t)BB * Cout * HW;
  if (idx >= total) return;
  int w  = (int)(idx % WWW);
  int h  = (int)((idx / WWW) % HHH);
  int oc = (int)((idx / HW) % Cout);
  int b  = (int)(idx / ((size_t)Cout * HW));
  int Cin = Cout / mult;
  const float* ip = in + ((size_t)b * Cin + oc / mult) * HW;
  const float* wp = wgt + (size_t)oc * 9;
  float s = 0.f;
#pragma unroll
  for (int dy = -1; dy <= 1; ++dy) {
    int hh = h + dy; if (hh < 0 || hh >= HHH) continue;
#pragma unroll
    for (int dx = -1; dx <= 1; ++dx) {
      int wc = w + dx; if (wc < 0 || wc >= WWW) continue;
      s += ip[hh * WWW + wc] * wp[(dy + 1) * 3 + (dx + 1)];
    }
  }
  if (relu) s = fmaxf(s, 0.f);
  out[idx] = s;
}

// ---------------- 3x3 avg (replicate pad) + highband split fused ----------------
__global__ void avg_hb_kernel(const float* __restrict__ y, float* __restrict__ l,
                              float* __restrict__ hb) {
  size_t idx = (size_t)blockIdx.x * blockDim.x + threadIdx.x;
  size_t total = (size_t)BB * HID4 * HW;
  if (idx >= total) return;
  int w = (int)(idx % WWW);
  int h = (int)((idx / WWW) % HHH);
  const float* ip = y + (idx - (size_t)(h * WWW + w));
  float s = 0.f;
#pragma unroll
  for (int dy = -1; dy <= 1; ++dy) {
    int hh = min(max(h + dy, 0), HHH - 1);
#pragma unroll
    for (int dx = -1; dx <= 1; ++dx) {
      int wc = min(max(w + dx, 0), WWW - 1);
      s += ip[hh * WWW + wc];
    }
  }
  s *= (1.f / 9.f);
  l[idx]  = s;
  hb[idx] = y[idx] - s;
}

// ---------------- ld/hd: groups=255, 4 in per group, virtual concat -------------
__global__ void mix_conv_kernel(const float* __restrict__ lf, const float* __restrict__ hbf,
                                const float* __restrict__ wgt, float* __restrict__ out,
                                int swap, int ocOff) {
  size_t idx = (size_t)blockIdx.x * blockDim.x + threadIdx.x;
  size_t total = (size_t)BB * HIDQ * HW;
  if (idx >= total) return;
  int w  = (int)(idx % WWW);
  int h  = (int)((idx / WWW) % HHH);
  int oc = (int)((idx / HW) % HIDQ);
  int b  = (int)(idx / ((size_t)HIDQ * HW));
  float s = 0.f;
  for (int j = 0; j < 4; ++j) {
    int ci = oc * 4 + j;
    int fromFirst = (ci < HID2) ? 1 : 0;
    const float* src = ((fromFirst ^ swap) != 0) ? lf : hbf;
    const float* ip = src + ((size_t)b * HID4 + ci) * HW;
    const float* wp = wgt + ((size_t)oc * 4 + j) * 9;
#pragma unroll
    for (int dy = -1; dy <= 1; ++dy) {
      int hh = h + dy; if (hh < 0 || hh >= HHH) continue;
#pragma unroll
      for (int dx = -1; dx <= 1; ++dx) {
        int wc = w + dx; if (wc < 0 || wc >= WWW) continue;
        s += ip[hh * WWW + wc] * wp[(dy + 1) * 3 + (dx + 1)];
      }
    }
  }
  s = fmaxf(s, 0.f);
  out[((size_t)b * HID2 + ocOff + oc) * HW + h * WWW + w] = s;
}

// ---------------- q/k row inverse L2 norms (384 rows of 16384) ------------------
__global__ void rownorm_kernel(const float* __restrict__ qkv, float* __restrict__ invn) {
  int row = blockIdx.x;                       // [b][qk][h][c]
  int c  = row % CHD;
  int h  = (row / CHD) % NHH;
  int qk = (row / (CHD * NHH)) % 2;
  int b  = row / (CHD * NHH * 2);
  const float* p = qkv + ((size_t)b * (3 * CC) + qk * CC + h * CHD + c) * HW;
  float s = 0.f;
  for (int i = threadIdx.x; i < HW; i += 256) { float v = p[i]; s += v * v; }
  __shared__ float red[256];
  red[threadIdx.x] = s;
  __syncthreads();
  for (int st = 128; st > 0; st >>= 1) {
    if (threadIdx.x < st) red[threadIdx.x] += red[threadIdx.x + st];
    __syncthreads();
  }
  if (threadIdx.x == 0) invn[row] = 1.f / fmaxf(sqrtf(red[0]), 1e-12f);
}

// ---------------- Gram matrix attn = (q kT) * rq * rk * tau via WMMA ------------
__global__ __launch_bounds__(256)
void attn_gram_kernel(const float* __restrict__ qkv, const float* __restrict__ invn,
                      const float* __restrict__ temp, float* __restrict__ attn) {
  int bh = blockIdx.x;
  int h = bh % NHH, b = bh / NHH;
  int lane = threadIdx.x & 31, wave = threadIdx.x >> 5;
  int half = lane >> 4, lr = lane & 15;
  int rc = min(lr, CHD - 1);                  // clamped row (branchless OOB)
  float rmask = (lr < CHD) ? 1.f : 0.f;
  const float* qb = qkv + ((size_t)b * (3 * CC) + h * CHD + rc) * HW;
  const float* kb = qkv + ((size_t)b * (3 * CC) + CC + h * CHD + rc) * HW;
  v8f acc = {0.f,0.f,0.f,0.f,0.f,0.f,0.f,0.f};
  int kstart = wave * (HW / 8);               // 8 waves split K = 16384
  for (int kt = 0; kt < (HW / 8) / 32; ++kt) {
    int kbase = kstart + kt * 32;
    v16s as, bs;
    int selA = half ? 8 : 0;
#pragma unroll
    for (int e = 0; e < 16; ++e) {
      int kk = ((e & 8) ? 16 : 0) + selA + (e & 7);
      as[e] = (short)f2bf(qb[kbase + kk] * rmask);
    }
    int selB = half ? 8 : 0;
#pragma unroll
    for (int e = 0; e < 16; ++e)
      bs[e] = (short)f2bf(kb[kbase + ((e & 8) ? 16 : 0) + selB + (e & 7)] * rmask);
    acc = wmma_bf16(as, bs, acc);
  }
  __shared__ float red[8][16][16];
#pragma unroll
  for (int r = 0; r < 8; ++r) red[wave][r + (half ? 8 : 0)][lr] = acc[r];
  __syncthreads();
  int t = threadIdx.x;
  int m = t >> 4, nn = t & 15;
  if (m < CHD && nn < CHD) {
    float s = 0.f;
    for (int wv = 0; wv < 8; ++wv) s += red[wv][m][nn];
    const float* iq = invn + ((b * 2 + 0) * NHH + h) * CHD;
    const float* ik = invn + ((b * 2 + 1) * NHH + h) * CHD;
    s *= iq[m] * ik[nn] * temp[h];
    attn[(((size_t)b * NHH + h) * CHD + m) * CHD + nn] = s;
  }
}

// ---------------- STS gating: 4 soft-shrinks + softmax + weighted sum -----------
__global__ __launch_bounds__(256)
void sts_kernel(const float* __restrict__ attn, const float* __restrict__ tw,
                const float* __restrict__ w1, const float* __restrict__ b1,
                const float* __restrict__ w2, const float* __restrict__ b2,
                float* __restrict__ attnc) {
  __shared__ float sA[BB * NHH * CHD * CHD];
  __shared__ float savg[BB][NHH];
  __shared__ float shid[4][BB][NHH];
  __shared__ float sthr[4][BB][NHH];
  int t = threadIdx.x;
  for (int i = t; i < BB * NHH * CHD * CHD; i += 256) sA[i] = attn[i];
  __syncthreads();
  if (t < BB * NHH) {
    int b = t / NHH, h = t % NHH;
    const float* p = sA + (b * NHH + h) * CHD * CHD;
    float s = 0.f;
    for (int i = 0; i < CHD * CHD; ++i) s += fabsf(p[i]);
    savg[b][h] = s / (float)(CHD * CHD);
  }
  __syncthreads();
  if (t < 4 * BB * NHH) {
    int j = t % NHH, b = (t / NHH) % BB, s = t / (NHH * BB);
    float a = b1[s * NHH + j];
    for (int i = 0; i < NHH; ++i) a += savg[b][i] * w1[(s * NHH + j) * NHH + i];
    shid[s][b][j] = fmaxf(a, 0.f);
  }
  __syncthreads();
  if (t < 4 * BB * NHH) {
    int j = t % NHH, b = (t / NHH) % BB, s = t / (NHH * BB);
    float a = b2[s * NHH + j];
    for (int i = 0; i < NHH; ++i) a += shid[s][b][i] * w2[(s * NHH + j) * NHH + i];
    float g = 1.f / (1.f + expf(-a));
    sthr[s][b][j] = savg[b][j] * g;
  }
  __syncthreads();
  if (t < BB * NHH * CHD) {
    int c = t % CHD, h = (t / CHD) % NHH, b = t / (CHD * NHH);
    const float* arow = sA + ((b * NHH + h) * CHD + c) * CHD;
    float out[CHD];
    for (int d = 0; d < CHD; ++d) out[d] = 0.f;
    for (int s = 0; s < 4; ++s) {
      float thr = sthr[s][b][h];
      float row[CHD]; float mx = -1e30f;
      for (int d = 0; d < CHD; ++d) {
        float a = arow[d];
        float v = copysignf(fmaxf(fabsf(a) - thr, 0.f), a);
        row[d] = v; mx = fmaxf(mx, v);
      }
      float se = 0.f;
      for (int d = 0; d < CHD; ++d) { row[d] = expf(row[d] - mx); se += row[d]; }
      float tws = tw[s] / se;
      for (int d = 0; d < CHD; ++d) out[d] += tws * row[d];
    }
    float* op = attnc + ((size_t)(b * NHH + h) * CHD + c) * CHD;
    for (int d = 0; d < CHD; ++d) op[d] = out[d];
  }
}

// ---------------- out = attn_c @ v  via WMMA (K=12 padded to 32) ----------------
__global__ __launch_bounds__(256)
void attn_apply_kernel(const float* __restrict__ attnc, const float* __restrict__ qkv,
                       float* __restrict__ out) {
  int bh = blockIdx.y;
  int h = bh % NHH, b = bh / NHH;
  int lane = threadIdx.x & 31, wave = threadIdx.x >> 5;
  int half = lane >> 4, lr = lane & 15;
  __shared__ float sAc[CHD * CHD];
  for (int i = threadIdx.x; i < CHD * CHD; i += 256)
    sAc[i] = attnc[(size_t)bh * CHD * CHD + i];
  __syncthreads();
  const float* vb = qkv + ((size_t)b * (3 * CC) + 2 * CC + h * CHD) * HW;
  int n0 = blockIdx.x * 128 + wave * 16;
  int rc = min(lr, CHD - 1);
  float rmask = (lr < CHD) ? 1.f : 0.f;
  v16s as, bs;
  int selA = half ? 8 : 0;
#pragma unroll
  for (int e = 0; e < 16; ++e) {
    int kk = ((e & 8) ? 16 : 0) + selA + (e & 7);
    int kc = min(kk, CHD - 1);
    float km = (kk < CHD) ? 1.f : 0.f;
    as[e] = (short)f2bf(sAc[rc * CHD + kc] * rmask * km);
  }
  int selB = half ? 8 : 0;
#pragma unroll
  for (int e = 0; e < 16; ++e) {
    int d  = ((e & 8) ? 16 : 0) + selB + (e & 7);
    int dc = min(d, CHD - 1);
    float dm = (d < CHD) ? 1.f : 0.f;
    bs[e] = (short)f2bf(vb[(size_t)dc * HW + n0 + lr] * dm);
  }
  v8f acc = {0.f,0.f,0.f,0.f,0.f,0.f,0.f,0.f};
  acc = wmma_bf16(as, bs, acc);
  int n = n0 + lr;
#pragma unroll
  for (int r = 0; r < 8; ++r) {
    int m = r + (half ? 8 : 0);
    if (m < CHD) out[((size_t)b * CC + h * CHD + m) * HW + n] = acc[r];
  }
}

// ============================= launch =============================
extern "C" void kernel_launch(void* const* d_in, const int* in_sizes, int n_in,
                              void* d_out, int out_size, void* d_ws, size_t ws_size,
                              hipStream_t stream) {
  (void)in_sizes; (void)n_in; (void)out_size; (void)ws_size;
  const float* x      = (const float*)d_in[0];
  const float* n1w    = (const float*)d_in[1];
  const float* n1b    = (const float*)d_in[2];
  const float* qkv_w  = (const float*)d_in[3];
  const float* qkvdw  = (const float*)d_in[4];
  const float* proj_w = (const float*)d_in[5];
  const float* temp   = (const float*)d_in[6];
  const float* tw     = (const float*)d_in[7];
  const float* aw1    = (const float*)d_in[8];
  const float* ab1    = (const float*)d_in[9];
  const float* aw2    = (const float*)d_in[10];
  const float* ab2    = (const float*)d_in[11];
  const float* n2w    = (const float*)d_in[12];
  const float* n2b    = (const float*)d_in[13];
  const float* in_w   = (const float*)d_in[14];
  const float* main_w = (const float*)d_in[15];
  const float* lp_w   = (const float*)d_in[16];
  const float* hp_w   = (const float*)d_in[17];
  const float* ld_w   = (const float*)d_in[18];
  const float* hd_w   = (const float*)d_in[19];
  const float* out_w  = (const float*)d_in[20];
  float* dout = (float*)d_out;

  // ---- workspace layout (floats), lifetime-overlapped ----
  float* p = (float*)d_ws;
  const size_t S96   = (size_t)BB * CC   * HW;
  const size_t S288  = (size_t)BB * 3*CC * HW;
  const size_t S510  = (size_t)BB * HID2 * HW;
  const size_t S1020 = (size_t)BB * HID4 * HW;
  float* f96    = p; p += S96;    // ln1 -> attn_out -> ln2
  float* f288a  = p; p += S288;   // qkv pre-dw
  float* f288b  = p; p += S288;   // qkv post-dw (q,k,v)
  float* f510   = p; p += S510;   // y -> ff(concat l2,h2)
  float* f1020a = p; p += S1020;  // main-out -> lf
  float* f1020b = p; p += S1020;  // l -> hbf
  float* f1020c = p; p += S1020;  // hb
  float* invn   = p; p += 512;
  float* attn   = p; p += (size_t)BB * NHH * CHD * CHD;
  float* attnc  = p; p += (size_t)BB * NHH * CHD * CHD;

  dim3 blk(256);
  // ---- attention branch ----
  ln_kernel<<<(BB * HW + 255) / 256, blk, 0, stream>>>(x, n1w, n1b, f96);
  wmma_gemm_kernel<<<dim3(HW / GBN, (3 * CC + GBM - 1) / GBM, BB), blk, 0, stream>>>(
      qkv_w, f96, nullptr, f288a, 3 * CC, CC, HW);
  dwconv_kernel<<<(unsigned)(((size_t)BB * 3 * CC * HW + 255) / 256), blk, 0, stream>>>(
      f288a, qkvdw, f288b, 3 * CC, 1, 0);
  rownorm_kernel<<<BB * 2 * NHH * CHD, blk, 0, stream>>>(f288b, invn);
  attn_gram_kernel<<<BB * NHH, blk, 0, stream>>>(f288b, invn, temp, attn);
  sts_kernel<<<1, blk, 0, stream>>>(attn, tw, aw1, ab1, aw2, ab2, attnc);
  attn_apply_kernel<<<dim3(HW / 128, BB * NHH), blk, 0, stream>>>(attnc, f288b, f96);
  wmma_gemm_kernel<<<dim3(HW / GBN, (CC + GBM - 1) / GBM, BB), blk, 0, stream>>>(
      proj_w, f96, x, dout, CC, CC, HW);          // fused residual -> d_out

  // ---- FFN branch ----
  ln_kernel<<<(BB * HW + 255) / 256, blk, 0, stream>>>(dout, n2w, n2b, f96);
  wmma_gemm_kernel<<<dim3(HW / GBN, (HID2 + GBM - 1) / GBM, BB), blk, 0, stream>>>(
      in_w, f96, nullptr, f510, HID2, CC, HW);
  dwconv_kernel<<<(unsigned)(((size_t)BB * HID4 * HW + 255) / 256), blk, 0, stream>>>(
      f510, main_w, f1020a, HID4, 2, 1);          // grouped, 2 out per group, relu
  avg_hb_kernel<<<(unsigned)(((size_t)BB * HID4 * HW + 255) / 256), blk, 0, stream>>>(
      f1020a, f1020b, f1020c);                    // l, hb
  dwconv_kernel<<<(unsigned)(((size_t)BB * HID4 * HW + 255) / 256), blk, 0, stream>>>(
      f1020b, lp_w, f1020a, HID4, 1, 1);          // lf (main-out buffer reused)
  dwconv_kernel<<<(unsigned)(((size_t)BB * HID4 * HW + 255) / 256), blk, 0, stream>>>(
      f1020c, hp_w, f1020b, HID4, 1, 1);          // hbf (l buffer reused)
  mix_conv_kernel<<<(unsigned)(((size_t)BB * HIDQ * HW + 255) / 256), blk, 0, stream>>>(
      f1020a, f1020b, ld_w, f510, 0, 0);          // l2 -> ff[0:255]
  mix_conv_kernel<<<(unsigned)(((size_t)BB * HIDQ * HW + 255) / 256), blk, 0, stream>>>(
      f1020a, f1020b, hd_w, f510, 1, HIDQ);       // h2 -> ff[255:510]
  wmma_gemm_kernel<<<dim3(HW / GBN, (CC + GBM - 1) / GBM, BB), blk, 0, stream>>>(
      out_w, f510, dout, dout, CC, HID2, HW);     // fused residual -> d_out
}
